// PoolingRetriever_26508538151379
// MI455X (gfx1250) — compile-verified, matrix operand order checked by reading, not run on hardware
//
#include <hip/hip_runtime.h>
#include <stdint.h>

// ---------------------------------------------------------------------------
// PoolingRetriever on MI455X (gfx1250).
// Softmax over a length-1 axis == 1.0, so the reference reduces exactly to
//   out = (x @ Wv^T + bv) @ Wo^T + bo      (rank-64 bottleneck GEMM chain)
// q_state/Wq/bq/Wk/bk are dead. HBM-bound: ~402MB traffic -> ~17us floor.
// bf16 WMMA keeps compute far under that floor; async global->LDS loads
// (ASYNCcnt) double-buffer the x stream so HBM and WMMA phases overlap.
// ---------------------------------------------------------------------------

typedef __attribute__((ext_vector_type(16))) __bf16 v16bf;
typedef __attribute__((ext_vector_type(2)))  __bf16 v2bf;
typedef __attribute__((ext_vector_type(8)))  float  v8f;

#define DIM     768
#define NKV     64          // H*HD
#define ROWS_PB 256         // rows of x per block (8 waves x 2 tiles x 16)
#define MT      2           // 16-row tiles per wave
#define KCH     32          // k-chunk of x staged in LDS per iteration

// LDS strides, padded for conflict-free WMMA fragment gathers.
// bf16 regions (stride in ushorts): fragment reads dword (k/2); rows 0..15
// plus a +4-dword shift for the upper half-wave -> 387/35-dword odd strides.
#define WV_STRIDE 774       // Wv  [64 ][768]  bf16
#define WO_STRIDE 70        // Wo  [768][64 ]  bf16 (row n = Wo[n][0..63])
#define TS_STRIDE 70        // t staging [256][64] bf16 (reuses x buffer 0)
// fp32 x chunk (stride in floats): 42 dwords -> b64-aligned rows (168 B),
// 42*m mod 64 distinct for m=0..15, upper-half +8 dword offset conflict-free.
#define XS_STRIDE 42        // x chunk [256][32] f32, double buffered

#define WV_BYTES (NKV * WV_STRIDE * 2)      //  99072
#define WO_BYTES (DIM * WO_STRIDE * 2)      // 107520
#define XB_BYTES (ROWS_PB * XS_STRIDE * 4)  //  43008 per buffer
#define LDS_BYTES (WV_BYTES + WO_BYTES + 2 * XB_BYTES)   // 292608 B

static_assert(LDS_BYTES <= 320 * 1024, "LDS over WGP budget");
static_assert(ROWS_PB * TS_STRIDE * 2 <= XB_BYTES, "t staging fits buffer 0");

#if __has_builtin(__builtin_amdgcn_global_load_async_to_lds_b64)
#define USE_ASYNC 1
#else
#define USE_ASYNC 0
#endif

#if USE_ASYNC
// Exact builtin pointee type per hipcc diagnostic:
//   '__attribute__((__vector_size__(2 * sizeof(int)))) int __device__ *'
typedef int v2i_vs __attribute__((vector_size(8)));
typedef __attribute__((address_space(1))) v2i_vs* g_b64_t;   // global src
typedef __attribute__((address_space(3))) v2i_vs* l_b64_t;   // LDS dst
#if __has_builtin(__builtin_amdgcn_s_wait_asynccnt)
#define WAIT_ASYNC(n) __builtin_amdgcn_s_wait_asynccnt(n)
#else
#define WAIT_ASYNC(n) asm volatile("s_wait_asynccnt %0" ::"i"(n) : "memory")
#endif
#else
#define WAIT_ASYNC(n)
#endif

static __device__ __forceinline__ uint16_t f2bf(float f) {
  uint32_t u = __float_as_uint(f);
  u += 0x7FFFu + ((u >> 16) & 1u);        // round-to-nearest-even
  return (uint16_t)(u >> 16);
}

// Pack two fp32 -> packed bf16 dword. Prefer the single-instruction
// v_cvt_pk_bf16_f32 when the toolchain exposes it; manual RNE otherwise.
static __device__ __forceinline__ uint32_t pk_bf16(float lo, float hi) {
#if __has_builtin(__builtin_amdgcn_cvt_pk_bf16_f32)
  union { v2bf v; uint32_t u; } c;
  c.v = __builtin_amdgcn_cvt_pk_bf16_f32(lo, hi);
  return c.u;
#else
  return (uint32_t)f2bf(lo) | ((uint32_t)f2bf(hi) << 16);
#endif
}

union FragAB { v16bf v; uint32_t u[8]; };

// bf16 A/B fragment gather (ISA 7.12.2, 16-bit 16x32): lane holds M/N=lane%16;
// pair v: K = (v>=4 ? 16 : 0) + 8*half + 2*(v&3) + {0,1} -> 8 ds_load_b32.
static __device__ __forceinline__ v16bf load_frag(const uint16_t* row_base,
                                                  int kbase, int half) {
  FragAB f;
  const uint32_t* p = (const uint32_t*)row_base;
  const int kb = kbase + half * 8;
#pragma unroll
  for (int v = 0; v < 8; ++v) {
    int kk = kb + ((v & 4) ? 16 : 0) + ((v & 3) << 1);
    f.u[v] = p[kk >> 1];
  }
  return f.v;
}

// A fragment from the fp32 x chunk: 8 contiguous float2 (ds_load_b64) + packed
// bf16 converts. Lets the chunk be filled by async loads, no VGPR round-trip.
static __device__ __forceinline__ v16bf load_fragA_f32(const float* row_base,
                                                       int half) {
  FragAB f;
  const int kb = half * 8;
#pragma unroll
  for (int v = 0; v < 8; ++v) {
    int kk = kb + ((v & 4) ? 16 : 0) + ((v & 3) << 1);
    float2 d = *(const float2*)(row_base + kk);
    f.u[v] = pk_bf16(d.x, d.y);
  }
  return f.v;
}

// Stage x[r0..r0+255][k0..k0+31] into an LDS fp32 buffer.
// Async path: exactly 16 global_load_async_to_lds_b64 per wave (so ASYNCcnt
// arithmetic in the pipeline is exact); rows clamped instead of predicated --
// duplicate reads of the last valid row are harmless, output stores are
// guarded. Coalesced 128B per 16 lanes.
static __device__ __forceinline__ void stage_chunk(const float* __restrict__ x,
                                                   float* buf, int r0, int k0,
                                                   int Bn, int tid) {
#pragma unroll
  for (int i = 0; i < 16; ++i) {
    int f   = i * 256 + tid;      // 4096 b64 transfers per chunk
    int row = f >> 4;             // 16 pairs per 32-float row
    int j   = f & 15;
    int rg  = r0 + row;
    if (rg >= Bn) rg = Bn - 1;    // clamp: keep issue count exact
    const float* gp = x + (size_t)rg * DIM + k0 + (j << 1);
    float*       lp = buf + row * XS_STRIDE + (j << 1);
#if USE_ASYNC
    __builtin_amdgcn_global_load_async_to_lds_b64(
        (g_b64_t)(void*)gp, (l_b64_t)(void*)lp, 0, 0);
#else
    *(float2*)lp = *(const float2*)gp;
#endif
  }
}

// ---------------- prep: fp32 weights -> bf16 workspace ---------------------
__global__ void pr_prep(const float* __restrict__ Wv, const float* __restrict__ Wo,
                        uint16_t* __restrict__ wvB, uint16_t* __restrict__ woB) {
  int i = blockIdx.x * blockDim.x + threadIdx.x;
  if (i < NKV * DIM) {
    wvB[i] = f2bf(Wv[i]);     // Wv[n][k]: n=0..63,  k=0..767
    woB[i] = f2bf(Wo[i]);     // Wo[n][k]: n=0..767, k=0..63
  }
}

// ---------------- main fused kernel ----------------------------------------
__global__ __launch_bounds__(256, 1)
void pr_main(const float* __restrict__ x,
             const uint16_t* __restrict__ wvB,
             const uint16_t* __restrict__ woB,
             const float* __restrict__ bv,
             const float* __restrict__ bo,
             float* __restrict__ out, int Bn) {
  extern __shared__ char smem[];
  uint16_t* wv  = (uint16_t*)smem;                           // [64][WV_STRIDE]
  uint16_t* wo  = (uint16_t*)(smem + WV_BYTES);              // [768][WO_STRIDE]
  float*    xb0 = (float*)(smem + WV_BYTES + WO_BYTES);      // [256][XS_STRIDE]
  float*    xb1 = (float*)(smem + WV_BYTES + WO_BYTES + XB_BYTES);
  uint16_t* tS  = (uint16_t*)xb0;                            // reused after GEMM1

  const int tid  = threadIdx.x;
  const int wave = tid >> 5;
  const int lane = tid & 31;
  const int l16  = lane & 15;
  const int half = lane >> 4;
  const int r0   = blockIdx.x * ROWS_PB;
  const int wrow0 = wave * (MT * 16);        // wave's first row within block

  // kick off the first x chunk; its HBM fetch overlaps the weight staging
  stage_chunk(x, xb0, r0, 0, Bn, tid);

  // ---- stage bf16 weights into LDS (u32 copies; L2-resident source) ----
  {
    const uint32_t* sv = (const uint32_t*)wvB;
    int row = tid >> 2;                      // 4 threads per 768-elem row
    int j0  = (tid & 3) * 96;                // 384 u32 per row
    uint32_t* dv = (uint32_t*)(wv + row * WV_STRIDE);
    const uint32_t* s = sv + row * 384 + j0;
#pragma unroll 8
    for (int j = 0; j < 96; ++j) dv[j0 + j] = s[j];

    const uint32_t* so = (const uint32_t*)woB;
    for (int rr = tid; rr < DIM; rr += 256) { // 64-elem rows, 32 u32 each
      uint32_t* d2 = (uint32_t*)(wo + rr * WO_STRIDE);
      const uint32_t* s2 = so + rr * 32;
#pragma unroll
      for (int j = 0; j < 32; ++j) d2[j] = s2[j];
    }
  }

  // ---- GEMM1: t = x @ Wv^T   (fp32 accum over K=768, software pipelined) --
  v8f acc[MT][4];
#pragma unroll
  for (int mt = 0; mt < MT; ++mt)
#pragma unroll
    for (int nt = 0; nt < 4; ++nt)
#pragma unroll
      for (int r = 0; r < 8; ++r) acc[mt][nt][r] = 0.0f;

  int cur = 0;
  for (int k0 = 0; k0 < DIM; k0 += KCH) {
    float* bcur = cur ? xb1 : xb0;
    float* bnxt = cur ? xb0 : xb1;
    if (k0 + KCH < DIM) {
      stage_chunk(x, bnxt, r0, k0 + KCH, Bn, tid);  // next chunk in flight
      WAIT_ASYNC(16);                               // current chunk drained
    } else {
      WAIT_ASYNC(0);
    }
    __syncthreads();                                // all waves' fills visible

    v16bf a[MT];
#pragma unroll
    for (int mt = 0; mt < MT; ++mt)
      a[mt] = load_fragA_f32(bcur + (wrow0 + mt * 16 + l16) * XS_STRIDE, half);
#pragma unroll
    for (int nt = 0; nt < 4; ++nt) {
      v16bf b = load_frag(wv + (nt * 16 + l16) * WV_STRIDE, k0, half);
#pragma unroll
      for (int mt = 0; mt < MT; ++mt)
        acc[mt][nt] = __builtin_amdgcn_wmma_f32_16x16x32_bf16(
            false, a[mt], false, b, (short)0, acc[mt][nt], false, false);
    }
    __syncthreads();                 // everyone done reading bcur before refill
    cur ^= 1;
  }

  // ---- +bv, round-trip t through wave-private LDS (C/D -> A layout) ----
  float bvv[4];
#pragma unroll
  for (int nt = 0; nt < 4; ++nt) bvv[nt] = bv[nt * 16 + l16];

#pragma unroll
  for (int mt = 0; mt < MT; ++mt)
#pragma unroll
    for (int nt = 0; nt < 4; ++nt)
#pragma unroll
      for (int r = 0; r < 8; ++r) {
        int trow = wrow0 + mt * 16 + r + half * 8;   // C/D: M = r + 8*half
        tS[trow * TS_STRIDE + nt * 16 + l16] = f2bf(acc[mt][nt][r] + bvv[nt]);
      }

  // wave-local producer/consumer through LDS: drain our DS stores
  asm volatile("s_wait_dscnt 0" ::: "memory");

  v16bf a2[MT][2];
#pragma unroll
  for (int mt = 0; mt < MT; ++mt)
#pragma unroll
    for (int c = 0; c < 2; ++c)
      a2[mt][c] = load_frag(tS + (wrow0 + mt * 16 + l16) * TS_STRIDE, c * 32, half);

  // ---- GEMM2: out = t @ Wo^T + bo  (K=64, 48 column tiles) ----
  for (int ntile = 0; ntile < DIM / 16; ++ntile) {
    v8f acc2[MT];
#pragma unroll
    for (int mt = 0; mt < MT; ++mt)
#pragma unroll
      for (int r = 0; r < 8; ++r) acc2[mt][r] = 0.0f;

#pragma unroll
    for (int c = 0; c < 2; ++c) {
      v16bf b = load_frag(wo + (ntile * 16 + l16) * WO_STRIDE, c * 32, half);
#pragma unroll
      for (int mt = 0; mt < MT; ++mt)
        acc2[mt] = __builtin_amdgcn_wmma_f32_16x16x32_bf16(
            false, a2[mt][c], false, b, (short)0, acc2[mt], false, false);
    }

    int   col = ntile * 16 + l16;
    float bov = bo[col];
#pragma unroll
    for (int mt = 0; mt < MT; ++mt)
#pragma unroll
      for (int r = 0; r < 8; ++r) {
        int rowg = r0 + wrow0 + mt * 16 + r + half * 8;
        if (rowg < Bn) out[(size_t)rowg * DIM + col] = acc2[mt][r] + bov;
      }
  }
}

// ---------------------------------------------------------------------------
extern "C" void kernel_launch(void* const* d_in, const int* in_sizes, int n_in,
                              void* d_out, int out_size, void* d_ws, size_t ws_size,
                              hipStream_t stream) {
  (void)n_in; (void)out_size; (void)ws_size;
  // setup_inputs order: x, q_state, Wq, bq, Wk, bk, Wv, bv, Wo, bo
  const float* x  = (const float*)d_in[0];
  const float* Wv = (const float*)d_in[6];
  const float* bv = (const float*)d_in[7];
  const float* Wo = (const float*)d_in[8];
  const float* bo = (const float*)d_in[9];
  float* out = (float*)d_out;
  const int Bn = in_sizes[0] / DIM;

  uint16_t* wvB = (uint16_t*)d_ws;                 // 64*768 bf16
  uint16_t* woB = wvB + NKV * DIM;                 // 768*64 bf16

  pr_prep<<<(NKV * DIM + 255) / 256, 256, 0, stream>>>(Wv, Wo, wvB, woB);

  (void)hipFuncSetAttribute((const void*)pr_main,
                            hipFuncAttributeMaxDynamicSharedMemorySize,
                            LDS_BYTES);
  const int grid = (Bn + ROWS_PB - 1) / ROWS_PB;   // 256 blocks for B=65536
  pr_main<<<grid, 256, LDS_BYTES, stream>>>(x, wvB, woB, bv, bo, out, Bn);
}
